// FullModel_57277683860075
// MI455X (gfx1250) — matrix operand order repaired and checked by reading, not compiled, not run on hardware
//
#include <hip/hip_runtime.h>
#include <cstdint>
#include <cstddef>

// ---------------------------------------------------------------------------
// MI455X (gfx1250, wave32) hypergraph-GNN forward pass.
// Dense GEMMs: v_wmma_f32_16x16x32_bf16 with f32 accumulate; weights are
// pre-packed once into B-fragment order (bf16). B fragments are prefetched
// 8 k-steps at a time (one load clause, one wait, 8 back-to-back WMMAs).
// Segment ops: f32 global atomics / order-encoded u32 atomicMin.
// GraphNorm: single pass (sum & sumsq identity). LayerNorm: block per row.
// ---------------------------------------------------------------------------

typedef __attribute__((ext_vector_type(16))) __bf16 v16bf;
typedef __attribute__((ext_vector_type(8)))  float  v8f;

__device__ __forceinline__ __bf16 f2bf(float f) {
  unsigned u = __builtin_bit_cast(unsigned, f);
  u += 0x7FFFu + ((u >> 16) & 1u);               // round to nearest even
  unsigned short hs = (unsigned short)(u >> 16);
  return __builtin_bit_cast(__bf16, hs);
}
__device__ __forceinline__ float lrelu_f(float v) { return v > 0.f ? v : 0.01f * v; }

// order-preserving float<->u32 key for atomic min
__device__ __forceinline__ unsigned fenc(float f) {
  unsigned u = __builtin_bit_cast(unsigned, f);
  return (u & 0x80000000u) ? ~u : (u | 0x80000000u);
}
__device__ __forceinline__ float fdec(unsigned k) {
  unsigned u = (k & 0x80000000u) ? (k ^ 0x80000000u) : ~k;
  return __builtin_bit_cast(float, u);
}

// ---------------------------------------------------------------------------
// Weight packer: W[Hout,K] f32 (row-major) -> Wp bf16 in B-fragment order:
//   Wp[((jt*KT + t)*32 + lane)*16 + e] = bf16( W[jt*16 + (lane&15),
//                                               t*32 + (lane>>4)*8 + (e&7) + (e>>3)*16] )
// so each lane of the GEMM loads its v16bf fragment as one contiguous 32B.
// ---------------------------------------------------------------------------
__global__ __launch_bounds__(256)
void pack_w_kernel(const float* __restrict__ W, __bf16* __restrict__ Wp,
                   int K, size_t total /* = Hout*K */)
{
  size_t idx = (size_t)blockIdx.x * 256 + threadIdx.x;
  if (idx >= total) return;
  const int KT   = K / 32;
  const int e    = (int)(idx & 15);
  const int lane = (int)((idx >> 4) & 31);
  const int t    = (int)((idx >> 9) % (size_t)KT);
  const int jt   = (int)(idx / ((size_t)KT * 512));
  const int row  = jt * 16 + (lane & 15);
  const int col  = t * 32 + (lane >> 4) * 8 + (e & 7) + (e >> 3) * 16;
  Wp[idx] = f2bf(W[(size_t)row * K + col]);
}

// ---------------------------------------------------------------------------
// WMMA GEMM:  C[R,Hout] = act( (rn? rn[r]*A[r,:] : A[r,:]) @ W^T + bias )
// Wp = pre-packed bf16 weights (B-fragment order). 8 waves per block,
// 16 rows per wave, A fragments preloaded in registers (K templated).
// bias must be a valid pointer (zero dummy for bias-less GEMMs).
// ---------------------------------------------------------------------------
template <int K>
__global__ __launch_bounds__(256)
void wmma_gemm_kernel(const float* __restrict__ A, const float* __restrict__ rn,
                      const __bf16* __restrict__ Wp, const float* __restrict__ bias,
                      float* __restrict__ C, int R, int Hout, int act)
{
  constexpr int KT = K / 32;
  const int lane = threadIdx.x & 31;
  const int wave = threadIdx.x >> 5;
  const int r0   = (blockIdx.x * 8 + wave) * 16;
  if (r0 >= R) return;
  const int  mrow  = lane & 15;        // A row / C column within tile
  const int  hf    = lane >> 4;        // lane half
  const int  arow  = r0 + mrow;
  const bool rowOK = arow < R;
  const float* Arow = A + (size_t)(rowOK ? arow : r0) * K;
  const float  scl  = rowOK ? (rn ? rn[arow] : 1.0f) : 0.0f;

  // A fragments: 16-bit A 16x32 layout; lane-half hf owns K = base..base+7 and
  // base+16..base+23 with base = 32t + 8*hf. Loads vectorized as float4.
  v16bf afrag[KT];
#pragma unroll
  for (int t = 0; t < KT; ++t) {
    const int base = t * 32 + hf * 8;           // multiple of 8 floats (32B)
    const float4 lo0 = *reinterpret_cast<const float4*>(Arow + base);
    const float4 lo1 = *reinterpret_cast<const float4*>(Arow + base + 4);
    const float4 hi0 = *reinterpret_cast<const float4*>(Arow + base + 16);
    const float4 hi1 = *reinterpret_cast<const float4*>(Arow + base + 20);
    v16bf a;
    a[0]  = f2bf(lo0.x * scl); a[1]  = f2bf(lo0.y * scl);
    a[2]  = f2bf(lo0.z * scl); a[3]  = f2bf(lo0.w * scl);
    a[4]  = f2bf(lo1.x * scl); a[5]  = f2bf(lo1.y * scl);
    a[6]  = f2bf(lo1.z * scl); a[7]  = f2bf(lo1.w * scl);
    a[8]  = f2bf(hi0.x * scl); a[9]  = f2bf(hi0.y * scl);
    a[10] = f2bf(hi0.z * scl); a[11] = f2bf(hi0.w * scl);
    a[12] = f2bf(hi1.x * scl); a[13] = f2bf(hi1.y * scl);
    a[14] = f2bf(hi1.z * scl); a[15] = f2bf(hi1.w * scl);
    afrag[t] = a;
  }

  const bool fullTile = (r0 + 16 <= R);
  const int jt = Hout >> 4;
  for (int j = 0; j < jt; ++j) {
    const int j0 = j * 16;
    const __bf16* Wj = Wp + ((size_t)j * KT * 32 + lane) * 16;
    const float bj = bias[j0 + mrow];            // unconditional load
    v8f acc = {0.f, 0.f, 0.f, 0.f, 0.f, 0.f, 0.f, 0.f};
    // Prefetch B fragments 8 k-steps at a time: 16 b128 loads issued as one
    // clause, single s_wait, then 8 back-to-back WMMAs.
#pragma unroll
    for (int c = 0; c < KT; c += 8) {
      v16bf bfrag[8];
#pragma unroll
      for (int t = 0; t < 8; ++t)
        bfrag[t] = *reinterpret_cast<const v16bf*>(Wj + (size_t)(c + t) * 512);
#pragma unroll
      for (int t = 0; t < 8; ++t)
        acc = __builtin_amdgcn_wmma_f32_16x16x32_bf16(false, afrag[c + t], false,
                                                      bfrag[t], (short)0, acc,
                                                      false, false);
    }
    float* Crow = C + (size_t)(r0 + hf * 8) * Hout + j0 + mrow;
    if (fullTile) {
#pragma unroll
      for (int rr = 0; rr < 8; ++rr) {          // M = rr + 8*hf, N = lane&15
        float v = acc[rr] + bj;
        if (act) v = lrelu_f(v);
        Crow[(size_t)rr * Hout] = v;
      }
    } else {
#pragma unroll
      for (int rr = 0; rr < 8; ++rr) {
        const int orow = r0 + rr + hf * 8;
        if (orow < R) {
          float v = acc[rr] + bj;
          if (act) v = lrelu_f(v);
          C[(size_t)orow * Hout + j0 + mrow] = v;
        }
      }
    }
  }
}

// ---------------------------------------------------------------------------
// Row L2 norm reciprocal: rn[r] = 1 / max(||X[r,:]||, 1e-12)
// ---------------------------------------------------------------------------
__global__ __launch_bounds__(256)
void rownorm_kernel(const float* __restrict__ X, int K, float* __restrict__ rn)
{
  __shared__ float sh[256];
  const int r = blockIdx.x;
  float s = 0.f;
  for (int j = threadIdx.x; j < K; j += 256) {
    float v = X[(size_t)r * K + j];
    s += v * v;
  }
  sh[threadIdx.x] = s; __syncthreads();
  for (int w = 128; w > 0; w >>= 1) {
    if (threadIdx.x < w) sh[threadIdx.x] += sh[threadIdx.x + w];
    __syncthreads();
  }
  if (threadIdx.x == 0) rn[r] = 1.f / fmaxf(sqrtf(sh[0]), 1e-12f);
}

// ---------------------------------------------------------------------------
// Row LayerNorm over width 256 (+optional second 256-wide source => 512).
// s1bits != null => first source is order-encoded min keys (decode inline).
// ---------------------------------------------------------------------------
__global__ __launch_bounds__(256)
void layernorm_kernel(const float* __restrict__ s1, const unsigned* __restrict__ s1bits,
                      const float* __restrict__ s2,
                      const float* __restrict__ g, const float* __restrict__ b,
                      float* __restrict__ Y, int R, int H2)
{
  __shared__ float sh[256];
  const int r = blockIdx.x;
  if (r >= R) return;
  const int j  = threadIdx.x;
  const int Ht = 256 + H2;
  float x0 = s1bits ? fdec(s1bits[(size_t)r * 256 + j]) : s1[(size_t)r * 256 + j];
  float x1 = (H2 > 0) ? s2[(size_t)r * H2 + j] : 0.f;
  sh[j] = x0 + x1; __syncthreads();
  for (int w = 128; w > 0; w >>= 1) { if (j < w) sh[j] += sh[j + w]; __syncthreads(); }
  const float mean = sh[0] / (float)Ht;
  __syncthreads();
  const float d0 = x0 - mean;
  const float d1 = (H2 > 0) ? (x1 - mean) : 0.f;
  sh[j] = d0 * d0 + d1 * d1; __syncthreads();
  for (int w = 128; w > 0; w >>= 1) { if (j < w) sh[j] += sh[j + w]; __syncthreads(); }
  const float rstd = rsqrtf(sh[0] / (float)Ht + 1e-5f);
  Y[(size_t)r * Ht + j] = d0 * rstd * g[j] + b[j];
  if (H2 > 0) Y[(size_t)r * Ht + 256 + j] = d1 * rstd * g[256 + j] + b[256 + j];
}

// ---------------------------------------------------------------------------
// GraphNorm column stats, single pass: per-column sum & sumsq partials, then
// finalize with var(x - mean*ms) = E[x^2] - mean^2*ms*(2-ms).
// ---------------------------------------------------------------------------
__global__ __launch_bounds__(256)
void colstats_partial_kernel(const float* __restrict__ X, int R, int chunk,
                             float* __restrict__ psum, float* __restrict__ psq)
{
  const int j  = threadIdx.x;
  const int p  = blockIdx.x;
  const int r0 = p * chunk;
  const int r1 = min(R, r0 + chunk);
  float s = 0.f, q = 0.f;
  for (int r = r0; r < r1; ++r) {
    float v = X[(size_t)r * 256 + j];
    s += v; q += v * v;
  }
  psum[(size_t)p * 256 + j] = s;
  psq [(size_t)p * 256 + j] = q;
}

__global__ __launch_bounds__(256)
void colstats_final_kernel(const float* __restrict__ psum, const float* __restrict__ psq,
                           int P, int R, const float* __restrict__ ms, float eps,
                           float* __restrict__ meanOut, float* __restrict__ invOut)
{
  const int j = threadIdx.x;
  float s = 0.f, q = 0.f;
  for (int p = 0; p < P; ++p) { s += psum[(size_t)p * 256 + j]; q += psq[(size_t)p * 256 + j]; }
  const float mean = s / (float)R;
  const float m2   = q / (float)R;
  const float msv  = ms[j];
  const float var  = m2 - mean * mean * msv * (2.f - msv);
  meanOut[j] = mean;
  invOut[j]  = rsqrtf(var + eps);
}

__global__ __launch_bounds__(256)
void gn_apply_kernel(const float* __restrict__ X, const float* __restrict__ mean,
                     const float* __restrict__ inv, const float* __restrict__ ms,
                     const float* __restrict__ w, const float* __restrict__ b,
                     float* __restrict__ Y, size_t n)
{
  size_t i = (size_t)blockIdx.x * 256 + threadIdx.x;
  if (i >= n) return;
  const int j = (int)(i & 255);
  Y[i] = w[j] * (X[i] - mean[j] * ms[j]) * inv[j] + b[j];
}

// ---------------------------------------------------------------------------
// Edge scatter kernels: one block per incidence, 256 features per block.
// ---------------------------------------------------------------------------
__global__ __launch_bounds__(256)
void scatter_add_kernel(const float* __restrict__ X, const int* __restrict__ gidx,
                        const int* __restrict__ sidx, const float* __restrict__ gscale,
                        float* __restrict__ out, int E)
{
  __shared__ int gS, sS;
  __shared__ float scS;
  const int e = blockIdx.x;
  if (e >= E) return;
  if (threadIdx.x == 0) {
    gS = gidx[e]; sS = sidx[e];
    scS = gscale ? gscale[gidx[e]] : 1.f;
  }
  __syncthreads();
  const int j = threadIdx.x;
  atomicAdd(&out[(size_t)sS * 256 + j], X[(size_t)gS * 256 + j] * scS);
}

__global__ __launch_bounds__(256)
void scatter_min_kernel(const float* __restrict__ X, const int* __restrict__ gidx,
                        const int* __restrict__ sidx, unsigned* __restrict__ bits, int E)
{
  __shared__ int gS, sS;
  const int e = blockIdx.x;
  if (e >= E) return;
  if (threadIdx.x == 0) { gS = gidx[e]; sS = sidx[e]; }
  __syncthreads();
  const int j = threadIdx.x;
  atomicMin(&bits[(size_t)sS * 256 + j], fenc(X[(size_t)gS * 256 + j]));
}

// ---------------------------------------------------------------------------
// Elementwise helpers
// ---------------------------------------------------------------------------
__global__ __launch_bounds__(256)
void conv_finalize_kernel(const float* __restrict__ acc, const float* __restrict__ invdeg,
                          const float* __restrict__ bias, const float* __restrict__ addb,
                          float* __restrict__ Y, size_t n)
{
  size_t i = (size_t)blockIdx.x * 256 + threadIdx.x;
  if (i >= n) return;
  const int j = (int)(i & 255);
  const size_t r = i >> 8;
  Y[i] = lrelu_f(acc[i] * invdeg[r] + bias[j]) + addb[i];
}

__global__ __launch_bounds__(256)
void add_decode_kernel(const float* __restrict__ xe, const unsigned* __restrict__ bits,
                       float* __restrict__ Y, size_t n)
{
  size_t i = (size_t)blockIdx.x * 256 + threadIdx.x;
  if (i >= n) return;
  Y[i] = xe[i] + fdec(bits[i]);
}

__global__ __launch_bounds__(256)
void fill_f32_kernel(float* __restrict__ p, float v, size_t n)
{
  size_t i = (size_t)blockIdx.x * 256 + threadIdx.x;
  if (i < n) p[i] = v;
}
__global__ __launch_bounds__(256)
void fill_u32_kernel(unsigned* __restrict__ p, unsigned v, size_t n)
{
  size_t i = (size_t)blockIdx.x * 256 + threadIdx.x;
  if (i < n) p[i] = v;
}

__global__ __launch_bounds__(256)
void edge_count_kernel(const int* __restrict__ src, const int* __restrict__ dst,
                       float* __restrict__ cn, float* __restrict__ cm, int E)
{
  int i = blockIdx.x * 256 + threadIdx.x;
  if (i >= E) return;
  atomicAdd(&cn[src[i]], 1.f);
  atomicAdd(&cm[dst[i]], 1.f);
}
__global__ __launch_bounds__(256)
void invert_kernel(float* __restrict__ p, size_t n)
{
  size_t i = (size_t)blockIdx.x * 256 + threadIdx.x;
  if (i >= n) return;
  float v = p[i];
  p[i] = (v > 0.f) ? 1.f / v : 0.f;
}

// ---------------------------------------------------------------------------
// Host orchestration
// ---------------------------------------------------------------------------
extern "C" void kernel_launch(void* const* d_in, const int* in_sizes, int n_in,
                              void* d_out, int out_size, void* d_ws, size_t ws_size,
                              hipStream_t stream)
{
  (void)n_in; (void)out_size; (void)ws_size;
  const int IN = 256, HH = 256;
  const int N = in_sizes[0] / IN;
  const int M = in_sizes[1] / IN;
  const int E = in_sizes[38] / 2;
  const int L = in_sizes[9] / HH;

  const float* x        = (const float*)d_in[0];
  const float* x_e      = (const float*)d_in[1];
  const float* x_struct = (const float*)d_in[2];
  const float* w_in  = (const float*)d_in[3];
  const float* b_in  = (const float*)d_in[4];
  const float* w_e   = (const float*)d_in[5];
  const float* b_e   = (const float*)d_in[6];
  const float* w_sem = (const float*)d_in[7];
  const float* b_sem = (const float*)d_in[8];
  const float* ln_n_g = (const float*)d_in[9];
  const float* ln_n_b = (const float*)d_in[10];
  const float* hg_w   = (const float*)d_in[11];
  const float* hg_b   = (const float*)d_in[12];
  const float* gn_s_w = (const float*)d_in[13];
  const float* gn_s_b = (const float*)d_in[14];
  const float* gn_s_ms= (const float*)d_in[15];
  const float* gn_l_w = (const float*)d_in[16];
  const float* gn_l_b = (const float*)d_in[17];
  const float* gn_l_ms= (const float*)d_in[18];
  const float* gn_d_w = (const float*)d_in[19];
  const float* gn_d_b = (const float*)d_in[20];
  const float* gn_d_ms= (const float*)d_in[21];
  const float* skip_w = (const float*)d_in[22];
  const float* skip_b = (const float*)d_in[23];
  const float* nf_ln1_g = (const float*)d_in[24];
  const float* nf_ln1_b = (const float*)d_in[25];
  const float* nf_w  = (const float*)d_in[26];
  const float* nf_b  = (const float*)d_in[27];
  const float* nf_ln2_g = (const float*)d_in[28];
  const float* nf_ln2_b = (const float*)d_in[29];
  const float* ef_ln1_g = (const float*)d_in[30];
  const float* ef_ln1_b = (const float*)d_in[31];
  const float* ef_w  = (const float*)d_in[32];
  const float* ef_b  = (const float*)d_in[33];
  const float* ef_ln2_g = (const float*)d_in[34];
  const float* ef_ln2_b = (const float*)d_in[35];
  const float* lin_w = (const float*)d_in[36];
  const float* lin_b = (const float*)d_in[37];
  const int*   ei    = (const int*)d_in[38];
  const int* srcI = ei;
  const int* dstI = ei + E;

  // ---- workspace carve (bump allocator) ----
  char* ws = (char*)d_ws;
  size_t off = 0;
  auto carve = [&](size_t bytes) -> void* {
    void* p = ws + off;
    off = (off + bytes + 255) & ~(size_t)255;
    return p;
  };
  const size_t NH = (size_t)N * HH, MH = (size_t)M * HH;
  float*    s_xs   = (float*)carve(NH * 4);
  float*    s_xn   = (float*)carve(NH * 4);
  float*    s_xe   = (float*)carve(MH * 4);
  float*    s_ln   = (float*)carve(NH * 4);
  float*    s_xw   = (float*)carve(NH * 4);
  float*    s_tmp2 = (float*)carve(NH * 4);
  float*    s_big  = (float*)carve(NH * 2 * 4);   // also aliases s_he|s_out
  float*    s_he   = s_big;
  float*    s_out  = s_big + NH;
  unsigned* s_agg  = (unsigned*)carve(MH * 4);
  float*    s_rn   = (float*)carve((size_t)N * 4);
  float*    inv_cn = (float*)carve((size_t)N * 4);
  float*    inv_cm = (float*)carve((size_t)M * 4);
  float*    s_zero = (float*)carve(256 * 4);      // zero bias dummy
  const int CHUNK = 512;
  const int Pmax  = (N + CHUNK - 1) / CHUNK;
  float* psum   = (float*)carve((size_t)Pmax * 256 * 4);
  float* psq    = (float*)carve((size_t)Pmax * 256 * 4);
  float* s_mean = (float*)carve(256 * 4);
  float* s_inv  = (float*)carve(256 * 4);

  // packed bf16 weights (B-fragment order)
  const size_t HW = (size_t)HH * HH;              // 65536
  __bf16* p_win  = (__bf16*)carve(HW * 2);
  __bf16* p_wsem = (__bf16*)carve(HW * 2);
  __bf16* p_we   = (__bf16*)carve(HW * 2);
  __bf16* p_hg   = (__bf16*)carve((size_t)L * 3 * HW * 2);
  __bf16* p_skip = (__bf16*)carve((size_t)L * 2 * HW * 2);
  __bf16* p_nf   = (__bf16*)carve(HW * 2 * 2);    // 256x512
  __bf16* p_ef   = (__bf16*)carve(HW * 2 * 2);    // 256x512
  __bf16* p_lin  = (__bf16*)carve(HW / 2 * 2);    // 128x256

  // ---- launch helpers ----
  auto gb = [](size_t n) { return dim3((unsigned)((n + 255) / 256)); };
  auto fillf = [&](float* p, float v, size_t n) {
    fill_f32_kernel<<<gb(n), 256, 0, stream>>>(p, v, n);
  };
  auto fillu = [&](unsigned* p, unsigned v, size_t n) {
    fill_u32_kernel<<<gb(n), 256, 0, stream>>>(p, v, n);
  };
  auto pack = [&](const float* W, __bf16* Wp, int K, int Hout) {
    size_t total = (size_t)Hout * K;
    pack_w_kernel<<<gb(total), 256, 0, stream>>>(W, Wp, K, total);
  };
  auto gemm = [&](const float* A, const float* rnp, const __bf16* Wp, const float* bias,
                  float* C, int R, int K, int Hout, int act) {
    dim3 grid((unsigned)((R + 127) / 128));
    const float* bp = bias ? bias : s_zero;
    if (K == 256)
      wmma_gemm_kernel<256><<<grid, 256, 0, stream>>>(A, rnp, Wp, bp, C, R, Hout, act);
    else
      wmma_gemm_kernel<512><<<grid, 256, 0, stream>>>(A, rnp, Wp, bp, C, R, Hout, act);
  };
  auto gnstats = [&](const float* X, int R, const float* ms, float eps) {
    int P = (R + CHUNK - 1) / CHUNK;
    colstats_partial_kernel<<<P, 256, 0, stream>>>(X, R, CHUNK, psum, psq);
    colstats_final_kernel<<<1, 256, 0, stream>>>(psum, psq, P, R, ms, eps, s_mean, s_inv);
  };
  auto gnapply = [&](const float* X, const float* ms, const float* w, const float* b,
                     float* Y, int R) {
    gn_apply_kernel<<<gb((size_t)R * 256), 256, 0, stream>>>(X, s_mean, s_inv, ms, w, b, Y,
                                                             (size_t)R * 256);
  };

  fillf(s_zero, 0.f, 256);

  // ---- pack all weight matrices (one-time, ~2.3 MB total, L2-resident) ----
  pack(w_in,  p_win,  256, 256);
  pack(w_sem, p_wsem, 256, 256);
  pack(w_e,   p_we,   256, 256);
  for (int c = 0; c < L * 3; ++c) pack(hg_w + (size_t)c * HW, p_hg + (size_t)c * HW, 256, 256);
  for (int c = 0; c < L * 2; ++c) pack(skip_w + (size_t)c * HW, p_skip + (size_t)c * HW, 256, 256);
  pack(nf_w, p_nf, 512, 256);
  pack(ef_w, p_ef, 512, 256);
  pack(lin_w, p_lin, 256, 128);

  // ---- degrees (node count cn over src, hyperedge count cm over dst) ----
  fillf(inv_cn, 0.f, N);
  fillf(inv_cm, 0.f, M);
  edge_count_kernel<<<gb(E), 256, 0, stream>>>(srcI, dstI, inv_cn, inv_cm, E);
  invert_kernel<<<gb(N), 256, 0, stream>>>(inv_cn, N);
  invert_kernel<<<gb(M), 256, 0, stream>>>(inv_cm, M);

  // ---- input projections: act(l2n(X) @ W^T + b) ----
  rownorm_kernel<<<N, 256, 0, stream>>>(x_struct, IN, s_rn);
  gemm(x_struct, s_rn, p_win, b_in, s_xs, N, 256, 256, 1);
  rownorm_kernel<<<N, 256, 0, stream>>>(x, IN, s_rn);
  gemm(x, s_rn, p_wsem, b_sem, s_xn, N, 256, 256, 1);
  rownorm_kernel<<<M, 256, 0, stream>>>(x_e, IN, s_rn);
  gemm(x_e, s_rn, p_we, b_e, s_xe, M, 256, 256, 1);

  for (int i = 0; i < L; ++i) {
    // ---- xs branch: xs = lrelu(hgconv(LN(xs))) + gn_struct(LN(xs)) ----
    layernorm_kernel<<<N, 256, 0, stream>>>(s_xs, nullptr, nullptr,
                                            ln_n_g + (size_t)i * HH, ln_n_b + (size_t)i * HH,
                                            s_ln, N, 0);
    gnstats(s_ln, N, gn_s_ms + (size_t)i * HH, 256.0f /* GN_STRUCT_EPS = H */);
    gnapply(s_ln, gn_s_ms + (size_t)i * HH, gn_s_w + (size_t)i * HH,
            gn_s_b + (size_t)i * HH, s_tmp2, N);
    gemm(s_ln, nullptr, p_hg + ((size_t)i * 3 + 0) * HW, nullptr, s_xw, N, 256, 256, 0);
    fillf(s_he, 0.f, MH);
    scatter_add_kernel<<<E, 256, 0, stream>>>(s_xw, srcI, dstI, nullptr, s_he, E);
    fillf(s_out, 0.f, NH);
    scatter_add_kernel<<<E, 256, 0, stream>>>(s_he, dstI, srcI, inv_cm, s_out, E);
    conv_finalize_kernel<<<gb(NH), 256, 0, stream>>>(s_out, inv_cn,
                                                     hg_b + ((size_t)i * 3 + 0) * HH,
                                                     s_tmp2, s_xs, NH);
    // ---- xn branch: xn = lrelu(hgconv(gn_llm(xn))) + skip(gn_llm(xn)) ----
    gnstats(s_xn, N, gn_l_ms + (size_t)i * HH, 1e-5f);
    gnapply(s_xn, gn_l_ms + (size_t)i * HH, gn_l_w + (size_t)i * HH,
            gn_l_b + (size_t)i * HH, s_ln, N);
    gemm(s_ln, nullptr, p_skip + ((size_t)i * 2 + 0) * HW,
         skip_b + ((size_t)i * 2 + 0) * HH, s_tmp2, N, 256, 256, 0);
    gemm(s_ln, nullptr, p_hg + ((size_t)i * 3 + 1) * HW, nullptr, s_xw, N, 256, 256, 0);
    fillf(s_he, 0.f, MH);
    scatter_add_kernel<<<E, 256, 0, stream>>>(s_xw, srcI, dstI, nullptr, s_he, E);
    fillf(s_out, 0.f, NH);
    scatter_add_kernel<<<E, 256, 0, stream>>>(s_he, dstI, srcI, inv_cm, s_out, E);
    conv_finalize_kernel<<<gb(NH), 256, 0, stream>>>(s_out, inv_cn,
                                                     hg_b + ((size_t)i * 3 + 1) * HH,
                                                     s_tmp2, s_xn, NH);
    // ---- agg = segment_min(xn[src] -> dst) ----
    fillu(s_agg, 0xFFFFFFFFu, MH);
    scatter_min_kernel<<<E, 256, 0, stream>>>(s_xn, srcI, dstI, s_agg, E);
    // ---- xe branch (dual hypergraph: src'=dst, dst'=src) ----
    add_decode_kernel<<<gb(MH), 256, 0, stream>>>(s_xe, s_agg, s_xw, MH);
    gnstats(s_xw, M, gn_d_ms + (size_t)i * HH, 1e-5f);
    gnapply(s_xw, gn_d_ms + (size_t)i * HH, gn_d_w + (size_t)i * HH,
            gn_d_b + (size_t)i * HH, s_ln, M);
    gemm(s_ln, nullptr, p_skip + ((size_t)i * 2 + 1) * HW,
         skip_b + ((size_t)i * 2 + 1) * HH, s_tmp2, M, 256, 256, 0);
    gemm(s_ln, nullptr, p_hg + ((size_t)i * 3 + 2) * HW, nullptr, s_xw, M, 256, 256, 0);
    fillf(s_he, 0.f, NH);   // dual "hyperedge" side has N slots
    scatter_add_kernel<<<E, 256, 0, stream>>>(s_xw, dstI, srcI, nullptr, s_he, E);
    fillf(s_out, 0.f, MH);
    scatter_add_kernel<<<E, 256, 0, stream>>>(s_he, srcI, dstI, inv_cn, s_out, E);
    conv_finalize_kernel<<<gb(MH), 256, 0, stream>>>(s_out, inv_cm,
                                                     hg_b + ((size_t)i * 3 + 2) * HH,
                                                     s_tmp2, s_xe, MH);
  }

  // ---- node fusion head ----
  layernorm_kernel<<<N, 256, 0, stream>>>(s_xs, nullptr, s_xn, nf_ln1_g, nf_ln1_b,
                                          s_big, N, 256);
  gemm(s_big, nullptr, p_nf, nf_b, s_xw, N, 512, 256, 1);
  layernorm_kernel<<<N, 256, 0, stream>>>(s_xw, nullptr, nullptr, nf_ln2_g, nf_ln2_b,
                                          s_ln, N, 0);
  fillu(s_agg, 0xFFFFFFFFu, MH);
  scatter_min_kernel<<<E, 256, 0, stream>>>(s_ln, srcI, dstI, s_agg, E);

  // ---- edge fusion head + output ----
  layernorm_kernel<<<M, 256, 0, stream>>>(nullptr, s_agg, s_xe, ef_ln1_g, ef_ln1_b,
                                          s_big, M, 256);
  gemm(s_big, nullptr, p_ef, ef_b, s_xw, M, 512, 256, 1);
  layernorm_kernel<<<M, 256, 0, stream>>>(s_xw, nullptr, nullptr, ef_ln2_g, ef_ln2_b,
                                          s_ln, M, 0);
  gemm(s_ln, nullptr, p_lin, lin_b, (float*)d_out, M, 256, 128, 0);
}
